// Engram_32564442038399
// MI455X (gfx1250) — matrix operand order repaired and checked by reading, not compiled, not run on hardware
//
#include <hip/hip_runtime.h>
#include <hip/hip_bf16.h>

// ---------------------------------------------------------------------------
// Engram scan for MI455X (gfx1250).
//   reference: sequential per-sample cosine-sim lookup + create/reinforce.
//   dominant work:  B*P*D = 2048*2048*1024 MACs  (protos @ p per step)
//   scheme: chunked delayed-update — per 16-sample chunk do one bf16 WMMA
//   GEMM  SIM[16x2048] = A(16x1024) x ProtoBf16^T  against a snapshot, then a
//   cheap exact-f32 sequential fixup/update pass.  All bank state, norms and
//   feedback output are exact f32; bf16 touches only argmax/threshold inputs.
// ---------------------------------------------------------------------------

#define DD        1024      // feature dim
#define PP        2048      // prototype slots
#define BB        2048      // batch
#define CHUNK     16        // samples per WMMA chunk (one M-tile)
#define NTHREADS  1024      // 32 wave32 waves
#define NWAVES    32
#define LRF       0.01f
#define SIM_TH    0.7f
#define NEW_TH    0.3f
#define EPSV      1e-8f

typedef __attribute__((ext_vector_type(16))) __bf16 v16bf;
typedef __attribute__((ext_vector_type(8)))  float  v8f;

struct VI { float v; int i; };

__device__ __forceinline__ VI viMax(VI a, VI b) {
  return (b.v > a.v || (b.v == a.v && b.i < a.i)) ? b : a;   // first-max ties
}
__device__ __forceinline__ VI viMin(VI a, VI b) {
  return (b.v < a.v || (b.v == a.v && b.i < a.i)) ? b : a;   // first-min ties
}

__device__ __forceinline__ float waveSum(float v) {
  #pragma unroll
  for (int o = 16; o > 0; o >>= 1) v += __shfl_xor(v, o, 32);
  return v;
}

// block-wide f32 sum over 1024 threads; red = 32-float LDS scratch
__device__ __forceinline__ float blockSum(float v, float* red, int tid) {
  int lane = tid & 31, w = tid >> 5;
  v = waveSum(v);
  __syncthreads();
  if (lane == 0) red[w] = v;
  __syncthreads();
  if (w == 0) {
    float x = red[lane];
    x = waveSum(x);
    if (lane == 0) red[0] = x;
  }
  __syncthreads();
  float r = red[0];
  __syncthreads();
  return r;
}

__device__ __forceinline__ VI blockMax(VI a, float* redV, int* redI, int tid) {
  int lane = tid & 31, w = tid >> 5;
  #pragma unroll
  for (int o = 16; o > 0; o >>= 1) {
    VI b; b.v = __shfl_xor(a.v, o, 32); b.i = __shfl_xor(a.i, o, 32);
    a = viMax(a, b);
  }
  __syncthreads();
  if (lane == 0) { redV[w] = a.v; redI[w] = a.i; }
  __syncthreads();
  if (w == 0) {
    VI x; x.v = redV[lane]; x.i = redI[lane];
    #pragma unroll
    for (int o = 16; o > 0; o >>= 1) {
      VI b; b.v = __shfl_xor(x.v, o, 32); b.i = __shfl_xor(x.i, o, 32);
      x = viMax(x, b);
    }
    if (lane == 0) { redV[0] = x.v; redI[0] = x.i; }
  }
  __syncthreads();
  VI r; r.v = redV[0]; r.i = redI[0];
  __syncthreads();
  return r;
}

__device__ __forceinline__ VI blockMin(VI a, float* redV, int* redI, int tid) {
  int lane = tid & 31, w = tid >> 5;
  #pragma unroll
  for (int o = 16; o > 0; o >>= 1) {
    VI b; b.v = __shfl_xor(a.v, o, 32); b.i = __shfl_xor(a.i, o, 32);
    a = viMin(a, b);
  }
  __syncthreads();
  if (lane == 0) { redV[w] = a.v; redI[w] = a.i; }
  __syncthreads();
  if (w == 0) {
    VI x; x.v = redV[lane]; x.i = redI[lane];
    #pragma unroll
    for (int o = 16; o > 0; o >>= 1) {
      VI b; b.v = __shfl_xor(x.v, o, 32); b.i = __shfl_xor(x.i, o, 32);
      x = viMin(x, b);
    }
    if (lane == 0) { redV[0] = x.v; redI[0] = x.i; }
  }
  __syncthreads();
  VI r; r.v = redV[0]; r.i = redI[0];
  __syncthreads();
  return r;
}

// ---------------------------------------------------------------------------
// Init: snapshot input bank into workspace (f32 + bf16) and compute row norms.
// Grid-wide (one block per prototype row) so the 8 MB copy runs at HBM rate.
// ---------------------------------------------------------------------------
__global__ __launch_bounds__(256) void engram_init(
    const float* __restrict__ protosIn,
    float* __restrict__ protoF32, __bf16* __restrict__ protoBf16,
    float* __restrict__ protoNorm) {
  int row = blockIdx.x;
  const float* src = protosIn + (size_t)row * DD;
  float*  dstF = protoF32  + (size_t)row * DD;
  __bf16* dstB = protoBf16 + (size_t)row * DD;
  float ss = 0.f;
  for (int d = threadIdx.x; d < DD; d += 256) {
    float v = src[d];
    dstF[d] = v;
    dstB[d] = (__bf16)v;
    ss += v * v;
  }
  ss = waveSum(ss);
  __shared__ float red[8];
  if ((threadIdx.x & 31) == 0) red[threadIdx.x >> 5] = ss;
  __syncthreads();
  if (threadIdx.x == 0) {
    float t = 0.f;
    #pragma unroll
    for (int i = 0; i < 8; ++i) t += red[i];
    protoNorm[row] = sqrtf(t);
  }
}

// ---------------------------------------------------------------------------
// Persistent scan kernel: one 1024-thread workgroup owns the whole scan.
// ---------------------------------------------------------------------------
__global__ __launch_bounds__(NTHREADS, 1) void engram_scan(
    const float* __restrict__ padrao, const float* __restrict__ erro,
    const float* __restrict__ forcaIn, const int* __restrict__ idadeIn,
    const int* __restrict__ nIn, const float* __restrict__ gainPtr,
    float* __restrict__ protoF32, __bf16* __restrict__ protoBf16,
    const float* __restrict__ protoNormG, float* __restrict__ out) {

  extern __shared__ char smemRaw[];
  float*  simSh   = (float*)smemRaw;                 // [CHUNK][PP]  128 KB
  float*  padSh   = simSh + CHUNK * PP;              // [CHUNK][DD]   64 KB
  __bf16* aSh     = (__bf16*)(padSh + CHUNK * DD);   // [CHUNK][DD]   32 KB
  float*  protnSh = (float*)(aSh + CHUNK * DD);      // [PP]           8 KB
  float*  forcaSh = protnSh + PP;                    // [PP]           8 KB
  int*    idadeSh = (int*)(forcaSh + PP);            // [PP]           8 KB
  float*  pnormSh = (float*)(idadeSh + PP);          // [CHUNK]
  float*  redSh   = pnormSh + CHUNK;                 // [NWAVES]
  int*    redISh  = (int*)(redSh + NWAVES);          // [NWAVES]
  int*    modSh   = redISh + NWAVES;                 // [1 + CHUNK]
  int*    nSh     = modSh + 1 + CHUNK;               // [1]

  const int tid  = threadIdx.x;
  const int lane = tid & 31;
  const int wave = tid >> 5;

  // ---- load persistent scalar state into LDS ----
  for (int i = tid; i < PP; i += NTHREADS) {
    forcaSh[i] = forcaIn[i];
    idadeSh[i] = idadeIn[i];
    protnSh[i] = protoNormG[i];
  }
  if (tid == 0) nSh[0] = nIn[0];
  const float gain = gainPtr[0];
  __syncthreads();

  for (int cb = 0; cb < BB; cb += CHUNK) {
    if (tid == 0) modSh[0] = 0;
    // ---- stage chunk of padrao: f32 (exact path) + bf16 A-operand ----
    for (int e = tid; e < CHUNK * DD; e += NTHREADS) {
      float v = padrao[(size_t)cb * DD + e];
      padSh[e] = v;
      aSh[e]   = (__bf16)v;
    }
    __syncthreads();
    // ---- sample norms (exact f32) ----
    for (int j = 0; j < CHUNK; ++j) {
      float v = padSh[j * DD + tid];
      float s = blockSum(v * v, redSh, tid);
      if (tid == 0) pnormSh[j] = sqrtf(s);
    }
    __syncthreads();

    // ---- WMMA GEMM: SIM = A(16xD) x ProtoBf16^T, normalized on store ----
    {
      const int hi   = lane >> 4;                  // half-wave select
      const int rowA = lane & 15;                  // A row (M)
      const int aoff = hi ? 8 : 0;                 // A K-offset per ISA layout
      const int boff = hi ? 16 : 0;                // unused; B loads use aoff
      for (int t = wave; t < PP / 16; t += NWAVES) {
        const int nb = t * 16;                     // N tile base (proto slots)
        const __bf16* bbase = protoBf16 + (size_t)(nb + (lane & 15)) * DD;
        v8f acc = {};
        for (int kb = 0; kb < DD; kb += 32) {
          // A fragment (LDS): lane<16 holds K 0-7,16-23; lane>=16 K 8-15,24-31
          v16bf a;
          ((uint4*)&a)[0] = *(const uint4*)(aSh + rowA * DD + kb + aoff);
          ((uint4*)&a)[1] = *(const uint4*)(aSh + rowA * DD + kb + 16 + aoff);
          // B fragment: lane = column (proto row); lane<16 holds K 0-7,16-23; lane>=16 K 8-15,24-31
          v16bf b;
          ((uint4*)&b)[0] = *(const uint4*)(bbase + kb + aoff);
          ((uint4*)&b)[1] = *(const uint4*)(bbase + kb + 16 + aoff);
          __builtin_prefetch(bbase + kb + 512, 0, 0);   // global_prefetch_b8
          acc = __builtin_amdgcn_wmma_f32_16x16x32_bf16(
              false, a, false, b, (short)0, acc, false, false);
        }
        // C/D layout: VGPR r -> M = r (lane<16) / r+8 (lane>=16), N = lane&15
        #pragma unroll
        for (int r = 0; r < 8; ++r) {
          int M = hi ? (r + 8) : r;
          int N = lane & 15;
          float pn = fmaxf(pnormSh[M], EPSV);
          float qn = fmaxf(protnSh[nb + N], EPSV);
          simSh[M * PP + nb + N] = acc[r] / (pn * qn);
        }
      }
    }
    __syncthreads();

    // ---- sequential per-sample pass with delayed-update fixups ----
    for (int j = 0; j < CHUNK; ++j) {
      // fixups: exact f32 re-dot against rows modified earlier in this chunk
      int mc = modSh[0];
      for (int m = 0; m < mc; ++m) {
        int r = modSh[1 + m];
        float d = blockSum(padSh[j * DD + tid] * protoF32[(size_t)r * DD + tid],
                           redSh, tid);
        if (tid == 0)
          simSh[j * PP + r] =
              d / (fmaxf(protnSh[r], EPSV) * fmaxf(pnormSh[j], EPSV));
      }
      __syncthreads();
      int ncur = nSh[0];

      // argmax over valid slots (ties -> first index, matching jnp.argmax)
      VI best; best.v = -__builtin_inff(); best.i = 0x7fffffff;
      for (int s = tid; s < PP; s += NTHREADS) {
        VI c; c.v = (s < ncur) ? simSh[j * PP + s] : -__builtin_inff(); c.i = s;
        best = viMax(best, c);
      }
      best = blockMax(best, redSh, redISh, tid);

      // uniform decision (every lane computes identically)
      const bool isEmpty  = (ncur == 0);
      const bool doReinf  = (!isEmpty) && (best.v >= SIM_TH);
      const float err     = erro[cb + j];
      const bool doCreate =
          isEmpty || ((!doReinf) && ((err > NEW_TH) || (ncur < PP)));
      const int idx = best.i;
      const float pj = padSh[j * DD + tid];
      float* orow = out + (size_t)(cb + j) * DD;

      if (doCreate) {
        int cidx;
        if (ncur >= PP) {   // overwrite weakest (ties -> first index)
          VI wk; wk.v = __builtin_inff(); wk.i = 0x7fffffff;
          for (int s = tid; s < PP; s += NTHREADS) {
            VI c; c.v = (s < ncur) ? forcaSh[s] : __builtin_inff(); c.i = s;
            wk = viMin(wk, c);
          }
          wk = blockMin(wk, redSh, redISh, tid);
          cidx = wk.i;
        } else {
          cidx = ncur;
        }
        protoF32[(size_t)cidx * DD + tid]  = pj;
        protoBf16[(size_t)cidx * DD + tid] = (__bf16)pj;
        orow[tid] = 0.f;
        __syncthreads();
        if (tid == 0) {
          protnSh[cidx] = pnormSh[j];          // same vector -> same norm
          forcaSh[cidx] = 1.f;
          idadeSh[cidx] = 0;
          if (ncur < PP) nSh[0] = ncur + 1;
          int m = modSh[0]; modSh[1 + m] = cidx; modSh[0] = m + 1;
        }
        __threadfence();
        __syncthreads();
      } else if (doReinf) {
        float old  = protoF32[(size_t)idx * DD + tid];
        float nv   = (1.f - LRF) * old + LRF * pj;
        float fnew = forcaSh[idx] + LRF;
        orow[tid] = (nv - pj) * fnew * gain;
        protoF32[(size_t)idx * DD + tid]  = nv;
        protoBf16[(size_t)idx * DD + tid] = (__bf16)nv;
        float s = blockSum(nv * nv, redSh, tid);   // exact new norm
        if (tid == 0) {
          protnSh[idx] = sqrtf(s);
          forcaSh[idx] = fnew;
          idadeSh[idx] += 1;
          int m = modSh[0]; modSh[1 + m] = idx; modSh[0] = m + 1;
        }
        __threadfence();
        __syncthreads();
      } else {  // no update: feedback from untouched prototype
        float old = protoF32[(size_t)idx * DD + tid];
        orow[tid] = (old - pj) * forcaSh[idx] * gain;
        __syncthreads();
      }
    }
    __threadfence();   // make bf16 bank writes visible to next chunk's GEMM
    __syncthreads();
  }
}

extern "C" void kernel_launch(void* const* d_in, const int* in_sizes, int n_in,
                              void* d_out, int out_size, void* d_ws,
                              size_t ws_size, hipStream_t stream) {
  const float* padrao   = (const float*)d_in[0];
  const float* erroP    = (const float*)d_in[1];
  const float* protosIn = (const float*)d_in[2];
  const float* forcaIn  = (const float*)d_in[3];
  const int*   idadeIn  = (const int*)d_in[4];
  const int*   nIn      = (const int*)d_in[5];
  const float* gainP    = (const float*)d_in[6];
  float* out = (float*)d_out;

  // workspace: f32 bank (8 MB) | bf16 bank (4 MB) | row norms (8 KB)
  char* ws = (char*)d_ws;
  float*  protoF32  = (float*)ws;
  __bf16* protoBf16 = (__bf16*)(ws + (size_t)PP * DD * 4);
  float*  protoNorm = (float*)(ws + (size_t)PP * DD * 4 + (size_t)PP * DD * 2);

  engram_init<<<PP, 256, 0, stream>>>(protosIn, protoF32, protoBf16, protoNorm);

  size_t smem = (size_t)4 * (CHUNK * PP)        // sim panel
              + (size_t)4 * (CHUNK * DD)        // f32 chunk
              + (size_t)2 * (CHUNK * DD)        // bf16 A operand
              + (size_t)4 * 3 * PP              // norms / forca / idade
              + 1024;                           // small scratch + padding
  engram_scan<<<1, NTHREADS, smem, stream>>>(padrao, erroP, forcaIn, idadeIn,
                                             nIn, gainP, protoF32, protoBf16,
                                             protoNorm, out);
}
